// GemmaTensorProductAttention_22900765622442
// MI455X (gfx1250) — compile-verified, matrix operand order checked by reading, not run on hardware
//
#include <hip/hip_runtime.h>
#include <hip/hip_bf16.h>

// ---------------------------------------------------------------------------
// GemmaTensorProductAttention for MI455X (gfx1250, wave32, WMMA + TDM).
// Pipeline: fp32->bf16 convert | fused projection GEMM (WMMA bf16, TDM A-tile)
// | rmsnorm+rope+rank-contract | flash attention (WMMA bf16, window+softcap) |
// output GEMM (WMMA bf16).
// All GEMM dims are exact tile multiples (M=4096, N in {1408,2048}, K=2048),
// so tile staging is branchless b128 traffic.
// ---------------------------------------------------------------------------

#define B_    2
#define S_    2048
#define HID_  2048
#define NH_   16
#define NKV_  8
#define D_    128
#define QR_   6
#define KR_   2
#define VR_   2
#define WIN_  1024
#define CAP_  50.0f
#define EPS_  1e-6f
#define NPROJ 1408            // 96 + 16 + 16 + 768 + 256 + 256

typedef __attribute__((ext_vector_type(16))) __bf16 v16bf;
typedef __attribute__((ext_vector_type(8)))  __bf16 v8bf;
typedef __attribute__((ext_vector_type(4)))  __bf16 v4bf;
typedef __attribute__((ext_vector_type(8)))  float  v8f;

union Frag16 { v16bf f; v8bf h[2]; };

__device__ __forceinline__ v8f wmma_bf16(const Frag16& a, const Frag16& b, v8f c) {
    return __builtin_amdgcn_wmma_f32_16x16x32_bf16(false, a.f, false, b.f,
                                                   (short)0, c, false, false);
}

__device__ __forceinline__ __bf16 bf16_lo(unsigned int w) {
    unsigned short u = (unsigned short)(w & 0xffffu);
    return __builtin_bit_cast(__bf16, u);
}
__device__ __forceinline__ __bf16 bf16_hi(unsigned int w) {
    unsigned short u = (unsigned short)(w >> 16);
    return __builtin_bit_cast(__bf16, u);
}

// ------------------------- Tensor Data Mover support -----------------------
// Probe-verified: __builtin_amdgcn_tensor_load_to_lds exists on both
// toolchains; 5 args on ROCm 7.2 (clang-22), 6 args on clang-23. Guarded so
// the host pass (x86) and TDM-less toolchains fall back to manual staging.

#if defined(__HIP_DEVICE_COMPILE__)
#if __has_builtin(__builtin_amdgcn_tensor_load_to_lds) && \
    __has_builtin(__builtin_amdgcn_s_wait_tensorcnt)
#define USE_TDM 1
#endif
#endif
#ifndef USE_TDM
#define USE_TDM 0
#endif

#if USE_TDM
typedef __attribute__((ext_vector_type(4))) unsigned int u32x4_t;
typedef __attribute__((ext_vector_type(4))) int          i32x4_t;
typedef __attribute__((ext_vector_type(8))) int          i32x8_t;

// Issue a 2D tile DMA: tile_d0 x tile_d1 bf16 elements, global row stride
// stride_elems, into LDS at lds_byte. D# packing per CDNA5 ISA (group0:
// count/lds/global/type; group1: data_size=2B, dims, tile dims, dim0 stride).
__device__ __forceinline__ void tdm_load_2d_bf16(const __bf16* gsrc,
                                                 unsigned lds_byte,
                                                 unsigned long long tensor_d0,
                                                 unsigned long long tensor_d1,
                                                 unsigned tile_d0, unsigned tile_d1,
                                                 unsigned long long stride_elems) {
    unsigned long long ga = (unsigned long long)(const void*)gsrc;
    u32x4_t g0 = { 1u,                                   // count=1, user D#
                   lds_byte,                             // lds_addr
                   (unsigned)(ga & 0xffffffffull),       // global_addr[31:0]
                   (unsigned)((ga >> 32) & 0x01ffffffu)  // global_addr[56:32]
                     | (2u << 30) };                     // type=2 ("image")
    i32x8_t g1;
    g1[0] = 0x00010000;                                  // data_size=1 (2B)
    g1[1] = (int)((tensor_d0 & 0xffffull) << 16);        // tensor_dim0[15:0]
    g1[2] = (int)(((tensor_d0 >> 16) & 0xffffull)        // tensor_dim0[31:16]
                  | ((tensor_d1 & 0xffffull) << 16));    // tensor_dim1[15:0]
    g1[3] = (int)(((tensor_d1 >> 16) & 0xffffull)        // tensor_dim1[31:16]
                  | ((unsigned long long)tile_d0 << 16));// tile_dim0
    g1[4] = (int)tile_d1;                                // tile_dim1 (dim2=0)
    g1[5] = (int)(stride_elems & 0xffffffffull);         // dim0_stride[31:0]
    g1[6] = (int)((stride_elems >> 32) & 0xffffull);     // dim0_stride[47:32]
    g1[7] = 0;
    i32x4_t z4 = { 0, 0, 0, 0 };
#if __clang_major__ >= 23
    i32x8_t z8 = { 0, 0, 0, 0, 0, 0, 0, 0 };
    __builtin_amdgcn_tensor_load_to_lds(g0, g1, z4, z4, z8, 0);
#else
    __builtin_amdgcn_tensor_load_to_lds(g0, g1, z4, z4, 0);
#endif
}
#endif // USE_TDM

// --------------------------- fp32 -> bf16 converts -------------------------

__global__ void k_cvt(const float* __restrict__ src, __bf16* __restrict__ dst,
                      int n4) {                       // n4 = n/4, n % 4 == 0
    int i = blockIdx.x * blockDim.x + threadIdx.x;
    if (i < n4) {
        const float4 v = *(const float4*)(src + (size_t)i * 4);
        v4bf o = { (__bf16)v.x, (__bf16)v.y, (__bf16)v.z, (__bf16)v.w };
        *(v4bf*)(dst + (size_t)i * 4) = o;
    }
}

// scatter a (HID x ncols) weight into the concatenated (HID x NPROJ) bf16 block
__global__ void k_cvt_cat(const float* __restrict__ src, __bf16* __restrict__ dst,
                          int ncols, int base) {
    int i = blockIdx.x * blockDim.x + threadIdx.x;
    int n = HID_ * ncols;
    if (i < n) {
        int k = i / ncols, c = i - k * ncols;
        dst[(size_t)k * NPROJ + base + c] = (__bf16)src[i];
    }
}

// ------------------------------- bf16 GEMM ---------------------------------
// C[M,N](f32) = A[M,K](bf16,row-major) * B[K,N](bf16,row-major)
// Block: 128x64 tile, 256 threads = 8 waves in a 4x2 grid of 32x32 wave tiles.
// Requires M%128==0, N%64==0, K%32==0 (true for all call sites).

#define GTM 128
#define GTN 64
#define GTK 32

__global__ __launch_bounds__(256)
void k_gemm_bf16(const __bf16* __restrict__ A, const __bf16* __restrict__ Bw,
                 float* __restrict__ C, int M, int N, int K) {
    __shared__ __attribute__((aligned(32))) __bf16 As[GTM * GTK];   // [m][k]
    __shared__ __attribute__((aligned(32))) __bf16 Bt[GTN * GTK];   // [n][k]

    const int tid   = threadIdx.x;
    const int lane  = tid & 31;
    const int w     = tid >> 5;
    const int wm    = w >> 1;       // 0..3
    const int wn    = w & 1;        // 0..1
    const int m0    = blockIdx.y * GTM;
    const int n0    = blockIdx.x * GTN;
    const int l16   = lane & 15;
    const int khalf = lane >> 4;

    // B staging coords: 256 chunks of 8 bf16 over the 32x64 tile
    const int bkk  = tid >> 3;      // 0..31 : k within tile
    const int bseg = tid & 7;       // 0..7  : 8-elem chunk along n

    v8f acc[2][2] = {};

    for (int kb = 0; kb < K; kb += GTK) {
        // ---- A tile: 128x32 bf16 -> LDS (TDM DMA, or b128 copies) ----
#if USE_TDM
        if (tid < 32) {     // one wave issues the descriptor (EXEC ignored)
            tdm_load_2d_bf16(A + (size_t)m0 * K + kb,
                             (unsigned)(size_t)(void*)As,
                             (unsigned long long)K, (unsigned long long)M,
                             GTK, GTM, (unsigned long long)K);
        }
#else
#pragma unroll
        for (int p = 0; p < 2; ++p) {
            int c   = tid + p * 256;            // 0..511 chunks of 8 bf16
            int row = c >> 2, seg = c & 3;
            const uint4 val =
                *(const uint4*)(A + (size_t)(m0 + row) * K + kb + seg * 8);
            *(uint4*)(As + row * GTK + seg * 8) = val;
        }
#endif
        // ---- B tile: 32x64 bf16, transposed into Bt[n][k] ----
        {
            const uint4 bv =
                *(const uint4*)(Bw + (size_t)(kb + bkk) * N + n0 + bseg * 8);
            unsigned int wd[4] = { bv.x, bv.y, bv.z, bv.w };
#pragma unroll
            for (int e = 0; e < 4; ++e) {
                Bt[(bseg * 8 + 2 * e)     * GTK + bkk] = bf16_lo(wd[e]);
                Bt[(bseg * 8 + 2 * e + 1) * GTK + bkk] = bf16_hi(wd[e]);
            }
        }
        // ---- prefetch next K-tile while this one is consumed ----
        if (kb + GTK < K) {
            __builtin_prefetch(A + (size_t)(m0 + (tid >> 1)) * K + kb + GTK
                                 + (tid & 1) * 16, 0, 1);
            __builtin_prefetch(Bw + (size_t)(kb + GTK + bkk) * N + n0 + bseg * 8,
                               0, 1);
        }
#if USE_TDM
        __builtin_amdgcn_s_wait_tensorcnt(0);   // no-op for waves 1..7
#endif
        __syncthreads();

        Frag16 af[2], bfv[2];
#pragma unroll
        for (int mi = 0; mi < 2; ++mi) {
            int m = wm * 32 + mi * 16 + l16;
            af[mi].h[0] = *(const v8bf*)(As + m * GTK + khalf * 8);
            af[mi].h[1] = *(const v8bf*)(As + m * GTK + 16 + khalf * 8);
        }
#pragma unroll
        for (int ni = 0; ni < 2; ++ni) {
            int n = wn * 32 + ni * 16 + l16;
            bfv[ni].h[0] = *(const v8bf*)(Bt + n * GTK + khalf * 8);
            bfv[ni].h[1] = *(const v8bf*)(Bt + n * GTK + 16 + khalf * 8);
        }
#pragma unroll
        for (int mi = 0; mi < 2; ++mi)
#pragma unroll
            for (int ni = 0; ni < 2; ++ni)
                acc[mi][ni] = wmma_bf16(af[mi], bfv[ni], acc[mi][ni]);
        __syncthreads();
    }

#pragma unroll
    for (int mi = 0; mi < 2; ++mi)
#pragma unroll
        for (int ni = 0; ni < 2; ++ni)
#pragma unroll
            for (int v = 0; v < 8; ++v) {
                int m = m0 + wm * 32 + mi * 16 + khalf * 8 + v;
                int n = n0 + wn * 32 + ni * 16 + l16;
                C[(size_t)m * N + n] = acc[mi][ni][v];
            }
}

// -------------------- rmsnorm + rope + rank contractions -------------------
// One 128-thread block per (b,s) row of proj.

__global__ __launch_bounds__(128)
void k_postproc(const float* __restrict__ proj,
                const float* __restrict__ cosb, const float* __restrict__ sinb,
                const float* __restrict__ qnw,  const float* __restrict__ knw,
                __bf16* __restrict__ Qb, __bf16* __restrict__ Kb,
                __bf16* __restrict__ Vb) {
    const int row = blockIdx.x;          // b*S + s
    const int b   = row / S_;
    const int s   = row - b * S_;
    const int d   = threadIdx.x;         // 0..127
    const float* pr = proj + (size_t)row * NPROJ;

    __shared__ float bq[QR_][D_], bk[KR_][D_], bv[VR_][D_];
    __shared__ float aq[NH_ * QR_], ak[NKV_ * KR_], av[NKV_ * VR_];
    __shared__ float red[4];

    if (d < 96) aq[d] = pr[d];
    if (d < 16) { ak[d] = pr[96 + d]; av[d] = pr[112 + d]; }

    // rmsnorm each rank-row of B_q / B_k
#pragma unroll 1
    for (int r = 0; r < QR_ + KR_; ++r) {
        const bool isq = r < QR_;
        const float* src = isq ? (pr + 128 + r * D_) : (pr + 896 + (r - QR_) * D_);
        float* dst = isq ? bq[r] : bk[r - QR_];
        const float* wv = isq ? qnw : knw;
        float x  = src[d];
        float ss = x * x;
#pragma unroll
        for (int off = 16; off; off >>= 1) ss += __shfl_xor(ss, off, 32);
        if ((d & 31) == 0) red[d >> 5] = ss;
        __syncthreads();
        float tot = red[0] + red[1] + red[2] + red[3];
        float inv = rsqrtf(tot * (1.0f / D_) + EPS_);
        dst[d] = x * inv * (1.0f + wv[d]);
        __syncthreads();
    }
    bv[0][d] = pr[1152 + d];
    bv[1][d] = pr[1280 + d];
    __syncthreads();

    // RoPE (pairs d, d+64)
    if (d < 64) {
        float c  = cosb[s * 64 + d];
        float si = sinb[s * 64 + d];
#pragma unroll
        for (int r = 0; r < QR_; ++r) {
            float x1 = bq[r][d], x2 = bq[r][d + 64];
            bq[r][d] = x1 * c - x2 * si;
            bq[r][d + 64] = x1 * si + x2 * c;
        }
#pragma unroll
        for (int r = 0; r < KR_; ++r) {
            float x1 = bk[r][d], x2 = bk[r][d + 64];
            bk[r][d] = x1 * c - x2 * si;
            bk[r][d + 64] = x1 * si + x2 * c;
        }
    }
    __syncthreads();

    const float qscale = 0.08838834764831845f / (float)QR_;   // D^-0.5 / QR
#pragma unroll 1
    for (int h = 0; h < NH_; ++h) {
        float acc = 0.0f;
#pragma unroll
        for (int r = 0; r < QR_; ++r) acc += aq[h * QR_ + r] * bq[r][d];
        Qb[(((size_t)b * NH_ + h) * S_ + s) * D_ + d] = (__bf16)(acc * qscale);
    }
#pragma unroll 1
    for (int kv = 0; kv < NKV_; ++kv) {
        float acck = 0.0f, accv = 0.0f;
#pragma unroll
        for (int r = 0; r < KR_; ++r) acck += ak[kv * KR_ + r] * bk[r][d];
#pragma unroll
        for (int r = 0; r < VR_; ++r) accv += av[kv * VR_ + r] * bv[r][d];
        Kb[(((size_t)b * NKV_ + kv) * S_ + s) * D_ + d] = (__bf16)(acck * (1.0f / KR_));
        Vb[(((size_t)b * NKV_ + kv) * S_ + s) * D_ + d] = (__bf16)(accv * (1.0f / VR_));
    }
}

// ------------------------------ flash attention ----------------------------
// One wave per (b,h,16-query tile); 32-key steps; windowed causal + softcap.

__global__ __launch_bounds__(32)
void k_attn(const __bf16* __restrict__ Qb, const __bf16* __restrict__ Kb,
            const __bf16* __restrict__ Vb, __bf16* __restrict__ Ob) {
    const int qtiles = S_ / 16;
    const int qt   = blockIdx.x % qtiles;
    const int h    = (blockIdx.x / qtiles) % NH_;
    const int b    = blockIdx.x / (qtiles * NH_);
    const int kv   = h >> 1;                 // NH/NKV = 2
    const int q0   = qt * 16;
    const int lane = threadIdx.x;
    const int l16  = lane & 15;
    const int khalf = lane >> 4;

    __shared__ __attribute__((aligned(32))) __bf16 Vt[D_ * 32];   // [d][key]
    __shared__ __attribute__((aligned(32))) __bf16 Pl[16 * 32];   // [m][key]

    const __bf16* Qp = Qb + (((size_t)b * NH_  + h ) * S_) * D_;
    const __bf16* Kp = Kb + (((size_t)b * NKV_ + kv) * S_) * D_;
    const __bf16* Vp = Vb + (((size_t)b * NKV_ + kv) * S_) * D_;

    // resident Q fragments (A-layout): 4 chunks of K=32 over D=128
    Frag16 qf[4];
#pragma unroll
    for (int dc = 0; dc < 4; ++dc) {
        const __bf16* src = Qp + (size_t)(q0 + l16) * D_ + dc * 32 + khalf * 8;
        qf[dc].h[0] = *(const v8bf*)src;
        qf[dc].h[1] = *(const v8bf*)(src + 16);
    }

    v8f o[8] = {};
    float mi[8], li[8];
#pragma unroll
    for (int v = 0; v < 8; ++v) { mi[v] = -1e30f; li[v] = 0.0f; }

    int kt0 = q0 - (WIN_ - 1);
    if (kt0 < 0) kt0 = 0;
    kt0 &= ~31;

    for (int kt = kt0; kt <= q0 + 15; kt += 32) {
        // ---- stage V tile (32 keys x 128 d) transposed into Vt[d][key] ----
#pragma unroll 4
        for (int p = 0; p < 16; ++p) {
            int c   = lane + p * 32;           // 0..511 chunks of 8 bf16
            int key = c >> 4, seg = c & 15;
            int kk  = kt + key; if (kk > S_ - 1) kk = S_ - 1;
            const uint4 wv = *(const uint4*)(Vp + (size_t)kk * D_ + seg * 8);
            unsigned int wd[4] = { wv.x, wv.y, wv.z, wv.w };
#pragma unroll
            for (int e = 0; e < 4; ++e) {
                Vt[(seg * 8 + 2 * e)     * 32 + key] = bf16_lo(wd[e]);
                Vt[(seg * 8 + 2 * e + 1) * 32 + key] = bf16_hi(wd[e]);
            }
        }
        // prefetch next key tile (K and V rows)
        if (kt + 32 <= q0 + 15) {
            int pk = kt + 32 + lane; if (pk > S_ - 1) pk = S_ - 1;
            __builtin_prefetch(Kp + (size_t)pk * D_, 0, 1);
            __builtin_prefetch(Vp + (size_t)pk * D_, 0, 1);
        }
        __syncthreads();

        // ---- scores: Q(16xD) . K^T(Dx32) as two 16-wide WMMA columns ----
        v8f sc0 = {}, sc1 = {};
        int key0 = kt + l16;      if (key0 > S_ - 1) key0 = S_ - 1;
        int key1 = kt + 16 + l16; if (key1 > S_ - 1) key1 = S_ - 1;
#pragma unroll
        for (int dc = 0; dc < 4; ++dc) {
            Frag16 kf0, kf1;
            const __bf16* s0p = Kp + (size_t)key0 * D_ + dc * 32 + khalf * 8;
            const __bf16* s1p = Kp + (size_t)key1 * D_ + dc * 32 + khalf * 8;
            kf0.h[0] = *(const v8bf*)s0p; kf0.h[1] = *(const v8bf*)(s0p + 16);
            kf1.h[0] = *(const v8bf*)s1p; kf1.h[1] = *(const v8bf*)(s1p + 16);
            sc0 = wmma_bf16(qf[dc], kf0, sc0);
            sc1 = wmma_bf16(qf[dc], kf1, sc1);
        }

        // ---- softcap + window mask + online softmax (rows in half-waves) --
        float p0[8], p1[8];
#pragma unroll
        for (int v = 0; v < 8; ++v) {
            int i  = q0 + khalf * 8 + v;
            int j0 = kt + l16;
            int j1 = kt + 16 + l16;
            float x0 = tanhf(sc0[v] * (1.0f / CAP_)) * CAP_;
            float x1 = tanhf(sc1[v] * (1.0f / CAP_)) * CAP_;
            bool a0 = (j0 <= i) && (j0 > i - WIN_);
            bool a1 = (j1 <= i) && (j1 > i - WIN_);
            float mx = fmaxf(a0 ? x0 : -1e30f, a1 ? x1 : -1e30f);
#pragma unroll
            for (int off = 1; off < 16; off <<= 1)
                mx = fmaxf(mx, __shfl_xor(mx, off, 16));
            float mnew = fmaxf(mi[v], mx);
            float fac  = __expf(mi[v] - mnew);
            float e0 = a0 ? __expf(x0 - mnew) : 0.0f;
            float e1 = a1 ? __expf(x1 - mnew) : 0.0f;
            float rs = e0 + e1;
#pragma unroll
            for (int off = 1; off < 16; off <<= 1)
                rs += __shfl_xor(rs, off, 16);
            li[v] = li[v] * fac + rs;
            mi[v] = mnew;
#pragma unroll
            for (int dc8 = 0; dc8 < 8; ++dc8) o[dc8][v] *= fac;
            p0[v] = e0;
            p1[v] = e1;
        }

        // ---- P: C-layout -> A-layout via LDS round trip ----
#pragma unroll
        for (int v = 0; v < 8; ++v) {
            Pl[(khalf * 8 + v) * 32 + l16]      = (__bf16)p0[v];
            Pl[(khalf * 8 + v) * 32 + 16 + l16] = (__bf16)p1[v];
        }
        __syncthreads();

        Frag16 pf;
        pf.h[0] = *(const v8bf*)(Pl + l16 * 32 + khalf * 8);
        pf.h[1] = *(const v8bf*)(Pl + l16 * 32 + 16 + khalf * 8);

        // ---- O += P(16x32) . V(32xD), 8 column chunks of 16 ----
#pragma unroll
        for (int dc8 = 0; dc8 < 8; ++dc8) {
            Frag16 vf;
            const __bf16* vp = Vt + (dc8 * 16 + l16) * 32 + khalf * 8;
            vf.h[0] = *(const v8bf*)vp;
            vf.h[1] = *(const v8bf*)(vp + 16);
            o[dc8] = wmma_bf16(pf, vf, o[dc8]);
        }
        __syncthreads();
    }

    // ---- finalize: O /= l, write bf16 (B,S,NH*D) for the output GEMM ----
#pragma unroll
    for (int dc8 = 0; dc8 < 8; ++dc8)
#pragma unroll
        for (int v = 0; v < 8; ++v) {
            float val = o[dc8][v] / li[v];
            int srow  = q0 + khalf * 8 + v;
            int col   = h * D_ + dc8 * 16 + l16;
            Ob[((size_t)b * S_ + srow) * (NH_ * D_) + col] = (__bf16)val;
        }
}

// ------------------------------- launcher ----------------------------------

extern "C" void kernel_launch(void* const* d_in, const int* in_sizes, int n_in,
                              void* d_out, int out_size, void* d_ws, size_t ws_size,
                              hipStream_t stream) {
    (void)in_sizes; (void)n_in; (void)out_size; (void)ws_size;
    const float* hs   = (const float*)d_in[0];
    const float* cosb = (const float*)d_in[1];
    const float* sinb = (const float*)d_in[2];
    const float* WAq  = (const float*)d_in[3];
    const float* WAk  = (const float*)d_in[4];
    const float* WAv  = (const float*)d_in[5];
    const float* WBq  = (const float*)d_in[6];
    const float* WBk  = (const float*)d_in[7];
    const float* WBv  = (const float*)d_in[8];
    const float* Wo   = (const float*)d_in[9];
    const float* qnw  = (const float*)d_in[10];
    const float* knw  = (const float*)d_in[11];
    // d_in[12..16]: caches + kv_write_indices — fully overwritten in the
    // reference and not part of the returned output; safe to ignore.
    float* out = (float*)d_out;

    const size_t MROWS = (size_t)B_ * S_;            // 4096
    char* cur = (char*)d_ws;
    auto take = [&](size_t bytes) {
        char* p = cur;
        cur += (bytes + 255) & ~(size_t)255;
        return p;
    };
    __bf16* hsb  = (__bf16*)take(MROWS * HID_ * 2);                 // 16.8 MB
    __bf16* Wcat = (__bf16*)take((size_t)HID_ * NPROJ * 2);         //  5.8 MB
    __bf16* Wob  = (__bf16*)take((size_t)HID_ * HID_ * 2);          //  8.4 MB
    float*  proj = (float*) take(MROWS * NPROJ * 4);                // 23.1 MB
    __bf16* Qb   = (__bf16*)take((size_t)B_ * NH_  * S_ * D_ * 2);  // 16.8 MB
    __bf16* Kb   = (__bf16*)take((size_t)B_ * NKV_ * S_ * D_ * 2);  //  8.4 MB
    __bf16* Vb   = (__bf16*)take((size_t)B_ * NKV_ * S_ * D_ * 2);  //  8.4 MB
    __bf16* Attn = (__bf16*)take(MROWS * (size_t)(NH_ * D_) * 2);   // 16.8 MB

    const int TPB = 256;
    // converts (all sizes are multiples of 4)
    {
        int n4 = (int)(MROWS * HID_ / 4);
        k_cvt<<<(n4 + TPB - 1) / TPB, TPB, 0, stream>>>(hs, hsb, n4);
    }
    {
        int n4 = HID_ * HID_ / 4;
        k_cvt<<<(n4 + TPB - 1) / TPB, TPB, 0, stream>>>(Wo, Wob, n4);
    }
    struct { const float* w; int ncols; int base; } wl[6] = {
        { WAq,  96,    0 }, { WAk,  16,   96 }, { WAv,  16,  112 },
        { WBq, 768,  128 }, { WBk, 256,  896 }, { WBv, 256, 1152 },
    };
    for (int i = 0; i < 6; ++i) {
        int n = HID_ * wl[i].ncols;
        k_cvt_cat<<<(n + TPB - 1) / TPB, TPB, 0, stream>>>(wl[i].w, Wcat,
                                                           wl[i].ncols, wl[i].base);
    }

    // fused projection GEMM: 4096 x 1408 x 2048
    {
        dim3 grid(NPROJ / GTN, (int)(MROWS / GTM));
        k_gemm_bf16<<<grid, 256, 0, stream>>>(hsb, Wcat, proj,
                                              (int)MROWS, NPROJ, HID_);
    }

    // rmsnorm + rope + rank contractions -> Q/K/V (bf16)
    k_postproc<<<(int)MROWS, 128, 0, stream>>>(proj, cosb, sinb, qnw, knw,
                                               Qb, Kb, Vb);

    // flash attention: one wave per (b, h, 16-query tile)
    k_attn<<<B_ * NH_ * (S_ / 16), 32, 0, stream>>>(Qb, Kb, Vb, Attn);

    // output projection: 4096 x 2048 x 2048
    {
        dim3 grid(HID_ / GTN, (int)(MROWS / GTM));
        k_gemm_bf16<<<grid, 256, 0, stream>>>(Attn, Wob, out,
                                              (int)MROWS, HID_, HID_);
    }
}
